// GINWithGlobal_4389456577285
// MI455X (gfx1250) — matrix operand order repaired and checked by reading, not compiled, not run on hardware
//
#include <hip/hip_runtime.h>
#include <hip/hip_bf16.h>

#define NODES 100000
#define EDGES 1600000

typedef __attribute__((ext_vector_type(16))) __bf16 v16bf;
typedef __attribute__((ext_vector_type(8)))  float  v8f;

// ---------------- weight cast + transpose: Wt[n][k] = (bf16)W[k][n] ----------
__global__ void k_castT(const float* __restrict__ W, __bf16* __restrict__ Wt,
                        int K, int N) {
  int i = blockIdx.x * blockDim.x + threadIdx.x;
  if (i >= K * N) return;
  int k = i / N, n = i - k * N;
  Wt[n * K + k] = (__bf16)W[i];
}

// ---------------- activation cast fp32 -> bf16 (vector x4) ------------------
__global__ void k_cast_act(const float* __restrict__ a, __bf16* __restrict__ o, long n) {
  long i = ((long)blockIdx.x * blockDim.x + threadIdx.x) * 4;
  if (i + 3 < n) {
    float4 v = *(const float4*)(a + i);
    o[i + 0] = (__bf16)v.x; o[i + 1] = (__bf16)v.y;
    o[i + 2] = (__bf16)v.z; o[i + 3] = (__bf16)v.w;
  }
}

// ---------------- agg = (1+eps) * x ----------------------------------------
__global__ void k_init_agg(const float* __restrict__ x, const float* __restrict__ epsp,
                           float* __restrict__ agg, long n) {
  long i = (long)blockIdx.x * blockDim.x + threadIdx.x;
  float s = 1.0f + *epsp;
  if (i < n) agg[i] = s * x[i];
}

// ---------------- edge scatter: agg[dst] += h[src] (one wave per edge) ------
__global__ void k_scatter(const float* __restrict__ h, const int* __restrict__ src,
                          const int* __restrict__ dst, float* __restrict__ agg, int F) {
  int lane = threadIdx.x & 31;
  int e = (blockIdx.x * blockDim.x + threadIdx.x) >> 5;
  if (e >= EDGES) return;
  int s = src[e], d = dst[e];
  const float* hp = h + (size_t)s * F;
  float* ap = agg + (size_t)d * F;
  for (int c = lane; c < F; c += 32)
    unsafeAtomicAdd(&ap[c], hp[c]);
}

// ---------------- WMMA GEMM: Out[M x 128] = A[M x K] @ W[K x 128] + bias ----
// A is bf16 row-major (M x K); Wt is bf16 N-major (Wt[n*K+k]).
// Workgroup = 8 waves = 64 rows x 128 cols. Staging via async global->LDS DMA.
__global__ __launch_bounds__(256) void k_gemm128(
    const __bf16* __restrict__ A, const __bf16* __restrict__ Wt,
    const float* __restrict__ bias, float* __restrict__ Out, int M, int K) {
  __shared__ __align__(32) __bf16 ldsW[128 * 128];  // 32 KB max
  __shared__ __align__(32) __bf16 ldsA[64 * 128];   // 16 KB max
  const int tid = threadIdx.x;
  const int row0 = blockIdx.x * 64;

  // LDS byte addresses (low 32 bits of the generic address are the LDS offset)
  unsigned ldsWb = (unsigned)(uintptr_t)(&ldsW[0]);
  unsigned ldsAb = (unsigned)(uintptr_t)(&ldsA[0]);

  // async-stage transposed weights: 128*K bf16 in 16B chunks
  const int wChunks = (128 * K) / 8;
  for (int i = tid; i < wChunks; i += 256) {
    unsigned la = ldsWb + (unsigned)i * 16u;
    unsigned va = (unsigned)i * 16u;
    asm volatile("global_load_async_to_lds_b128 %0, %1, %2 offset:0"
                 :: "v"(la), "v"(va), "s"(Wt) : "memory");
  }
  // async-stage A tile: 64*K bf16 in 16B chunks (tail rows masked at store)
  const __bf16* Ag = A + (size_t)row0 * K;
  const int aChunks = (64 * K) / 8;
  for (int i = tid; i < aChunks; i += 256) {
    unsigned la = ldsAb + (unsigned)i * 16u;
    unsigned va = (unsigned)i * 16u;
    asm volatile("global_load_async_to_lds_b128 %0, %1, %2 offset:0"
                 :: "v"(la), "v"(va), "s"(Ag) : "memory");
  }
  asm volatile("s_wait_asynccnt 0" ::: "memory");
  __syncthreads();

  const int wave = tid >> 5, lane = tid & 31;
  const int mt = wave & 3;    // 4 m-tiles of 16 rows
  const int nh = wave >> 2;   // 2 n-halves of 64 cols
  const int l16 = lane & 15, hi = lane >> 4;
  const int m = mt * 16 + l16;

  v8f acc[4] = {};
  for (int kb = 0; kb < K; kb += 32) {
    // A fragment: lanes 0-15 hold K=kb+0..15; lanes 16-31 hold K=kb+16..31
    union { uint4 q[2]; v16bf v; } ua;
    const __bf16* ap = &ldsA[m * K + kb + hi * 16];
    ua.q[0] = *(const uint4*)(ap);
    ua.q[1] = *(const uint4*)(ap + 8);
    v16bf a = ua.v;
    // B fragments: contiguous 32B per lane thanks to N-major staging
    const __bf16* bp = &ldsW[kb + hi * 16];
    const int nbase = nh * 64 + l16;
#pragma unroll
    for (int nt = 0; nt < 4; ++nt) {
      v16bf b = *(const v16bf*)(bp + (size_t)(nbase + nt * 16) * K);
      acc[nt] = __builtin_amdgcn_wmma_f32_16x16x32_bf16(
          false, a, false, b, (short)0, acc[nt], false, false);
    }
  }

  const int outm0 = row0 + mt * 16 + hi * 8;
#pragma unroll
  for (int nt = 0; nt < 4; ++nt) {
    int col = nh * 64 + nt * 16 + l16;
    float bv = bias[col];
#pragma unroll
    for (int r = 0; r < 8; ++r) {
      int gm = outm0 + r;
      if (gm < M) Out[(size_t)gm * 128 + col] = acc[nt][r] + bv;
    }
  }
}

// ---------------- zero a float buffer ---------------------------------------
__global__ void k_zero(float* __restrict__ p, long n) {
  long i = (long)blockIdx.x * blockDim.x + threadIdx.x;
  if (i < n) p[i] = 0.0f;
}

// ---------------- BN stats over 100k rows x 128 cols ------------------------
__global__ __launch_bounds__(256) void k_bn_stats(const float* __restrict__ h,
                                                  float* __restrict__ stats, int M) {
  __shared__ float s1[256], s2[256];
  int tid = threadIdx.x;
  int col = tid & 127, rof = tid >> 7;
  int rbase = blockIdx.x * 512;
  float a = 0.f, b = 0.f;
  for (int r = rof; r < 512; r += 2) {
    int row = rbase + r;
    if (row < M) { float v = h[(size_t)row * 128 + col]; a += v; b += v * v; }
  }
  s1[tid] = a; s2[tid] = b;
  __syncthreads();
  if (tid < 128) {
    unsafeAtomicAdd(&stats[col], s1[tid] + s1[tid + 128]);
    unsafeAtomicAdd(&stats[128 + col], s2[tid] + s2[tid + 128]);
  }
}

// ---------------- BN apply + ReLU (in place) --------------------------------
__global__ void k_bn_apply(float* __restrict__ h, const float* __restrict__ stats,
                           const float* __restrict__ gamma, const float* __restrict__ beta,
                           long n, float invM) {
  long i = (long)blockIdx.x * blockDim.x + threadIdx.x;
  if (i >= n) return;
  int col = (int)(i & 127);
  float m = stats[col] * invM;
  float var = stats[128 + col] * invM - m * m;
  float y = gamma[col] * (h[i] - m) * rsqrtf(var + 1e-5f) + beta[col];
  h[i] = fmaxf(y, 0.0f);
}

// ---------------- mean-pool (atomic) ----------------------------------------
__global__ void k_pool(const float* __restrict__ h, const int* __restrict__ batch,
                       float* __restrict__ pooled, float* __restrict__ cnt) {
  int t = blockIdx.x * blockDim.x + threadIdx.x;
  int node = t >> 5, q = t & 31;
  if (node >= NODES) return;
  int g = batch[node];
  float4 v = *(const float4*)(h + (size_t)node * 128 + q * 4);
  float* pp = pooled + (size_t)g * 128 + q * 4;
  unsafeAtomicAdd(pp + 0, v.x);
  unsafeAtomicAdd(pp + 1, v.y);
  unsafeAtomicAdd(pp + 2, v.z);
  unsafeAtomicAdd(pp + 3, v.w);
  if (q == 0) unsafeAtomicAdd(&cnt[g], 1.0f);
}

// ---------------- comb = [pooled/cnt , global_feat] -------------------------
__global__ void k_comb(const float* __restrict__ pooled, const float* __restrict__ cnt,
                       const float* __restrict__ gf, float* __restrict__ comb) {
  int i = blockIdx.x * blockDim.x + threadIdx.x;
  if (i >= 512 * 160) return;
  int g = i / 160, c = i - g * 160;
  comb[i] = (c < 128) ? pooled[g * 128 + c] / fmaxf(cnt[g], 1.0f)
                      : gf[g * 32 + (c - 128)];
}

// ---------------- small fp32 dense: out = A@W + b ---------------------------
__global__ void k_dense(const float* __restrict__ A, const float* __restrict__ W,
                        const float* __restrict__ b, float* __restrict__ out,
                        int M, int K, int N) {
  int i = blockIdx.x * blockDim.x + threadIdx.x;
  if (i >= M * N) return;
  int m = i / N, n = i - m * N;
  float acc = b[n];
  for (int k = 0; k < K; ++k) acc += A[m * K + k] * W[k * N + n];
  out[i] = acc;
}

// ---------------- fused per-column BN + ReLU (one block per column) ---------
__global__ __launch_bounds__(256) void k_bn_col(float* __restrict__ x,
                                                const float* __restrict__ gamma,
                                                const float* __restrict__ beta,
                                                int R, int N) {
  __shared__ float s1[256], s2[256];
  int col = blockIdx.x, tid = threadIdx.x;
  float a = 0.f, b = 0.f;
  for (int r = tid; r < R; r += 256) {
    float v = x[(size_t)r * N + col]; a += v; b += v * v;
  }
  s1[tid] = a; s2[tid] = b;
  __syncthreads();
  for (int o = 128; o > 0; o >>= 1) {
    if (tid < o) { s1[tid] += s1[tid + o]; s2[tid] += s2[tid + o]; }
    __syncthreads();
  }
  float m = s1[0] / R;
  float inv = rsqrtf(s2[0] / R - m * m + 1e-5f);
  float g = gamma[col], be = beta[col];
  for (int r = tid; r < R; r += 256) {
    size_t idx = (size_t)r * N + col;
    x[idx] = fmaxf(g * (x[idx] - m) * inv + be, 0.0f);
  }
}

// ---------------- final head: out[g] = z2[g] . Wm3 + bm3 --------------------
__global__ void k_out(const float* __restrict__ z, const float* __restrict__ W,
                      const float* __restrict__ b, float* __restrict__ out) {
  int g = blockIdx.x * blockDim.x + threadIdx.x;
  if (g >= 512) return;
  float acc = b[0];
  for (int k = 0; k < 128; ++k) acc += z[(size_t)g * 128 + k] * W[k];
  out[g] = acc;
}

extern "C" void kernel_launch(void* const* d_in, const int* in_sizes, int n_in,
                              void* d_out, int out_size, void* d_ws, size_t ws_size,
                              hipStream_t stream) {
  (void)in_sizes; (void)n_in; (void)out_size; (void)ws_size;
  const float* x      = (const float*)d_in[0];
  const int*   ei     = (const int*)d_in[1];     // [0..E) = src, [E..2E) = dst
  const int*   batch  = (const int*)d_in[2];
  const float* gfeat  = (const float*)d_in[3];
  const float* W0     = (const float*)d_in[4];
  const float* b0     = (const float*)d_in[5];
  const float* eps0   = (const float*)d_in[6];
  const float* gamma0 = (const float*)d_in[7];
  const float* beta0  = (const float*)d_in[8];
  const float* Wsk    = (const float*)d_in[9];
  const float* bsk    = (const float*)d_in[10];
  const float* epss   = (const float*)d_in[11];
  const float* gammas = (const float*)d_in[12];
  const float* betas  = (const float*)d_in[13];
  const float* Wm1    = (const float*)d_in[14];
  const float* bm1    = (const float*)d_in[15];
  const float* gm1    = (const float*)d_in[16];
  const float* btm1   = (const float*)d_in[17];
  const float* Wm2    = (const float*)d_in[18];
  const float* bm2    = (const float*)d_in[19];
  const float* gm2    = (const float*)d_in[20];
  const float* btm2   = (const float*)d_in[21];
  const float* Wm3    = (const float*)d_in[22];
  const float* bm3    = (const float*)d_in[23];
  float* out = (float*)d_out;

  char* wsb = (char*)d_ws;
  size_t off = 0;
  auto alloc = [&](size_t bytes) {
    void* p = wsb + off;
    off = (off + bytes + 255) & ~(size_t)255;
    return p;
  };
  const int NPAD = ((NODES + 63) / 64) * 64;        // 100032 (async-tile safe)
  float*  agg    = (float*)alloc((size_t)NODES * 128 * 4);
  float*  h      = (float*)alloc((size_t)NODES * 128 * 4);
  __bf16* aggbf  = (__bf16*)alloc((size_t)NPAD * 128 * 2);
  __bf16* Wt0    = (__bf16*)alloc(64 * 128 * 2);
  __bf16* Wt1    = (__bf16*)alloc(3 * 128 * 128 * 2);
  float*  stats  = (float*)alloc(256 * 4);
  float*  pooled = (float*)alloc(512 * 128 * 4);
  float*  cnt    = (float*)alloc(512 * 4);
  float*  comb   = (float*)alloc(512 * 160 * 4);
  float*  z1     = (float*)alloc(512 * 256 * 4);
  float*  z2     = (float*)alloc(512 * 128 * 4);

  // weights -> bf16, transposed (N-major)
  k_castT<<<(64 * 128 + 255) / 256, 256, 0, stream>>>(W0, Wt0, 64, 128);
  for (int l = 0; l < 3; ++l)
    k_castT<<<(128 * 128 + 255) / 256, 256, 0, stream>>>(
        Wsk + (size_t)l * 128 * 128, Wt1 + (size_t)l * 128 * 128, 128, 128);

  const long nh = (long)NODES * 128;
  const float invM = 1.0f / (float)NODES;

  // ---- layer 0 (64 -> 128) ----
  {
    long n0 = (long)NODES * 64;
    k_init_agg<<<(unsigned)((n0 + 255) / 256), 256, 0, stream>>>(x, eps0, agg, n0);
    k_scatter<<<EDGES / 8, 256, 0, stream>>>(x, ei, ei + EDGES, agg, 64);
    k_cast_act<<<(unsigned)(n0 / 4 / 256), 256, 0, stream>>>(agg, aggbf, n0);
    k_gemm128<<<(NODES + 63) / 64, 256, 0, stream>>>(aggbf, Wt0, b0, h, NODES, 64);
    k_zero<<<1, 256, 0, stream>>>(stats, 256);
    k_bn_stats<<<(NODES + 511) / 512, 256, 0, stream>>>(h, stats, NODES);
    k_bn_apply<<<(unsigned)((nh + 255) / 256), 256, 0, stream>>>(h, stats, gamma0, beta0, nh, invM);
  }

  // ---- layers 1..3 (128 -> 128) ----
  for (int l = 0; l < 3; ++l) {
    k_init_agg<<<(unsigned)((nh + 255) / 256), 256, 0, stream>>>(h, epss + l, agg, nh);
    k_scatter<<<EDGES / 8, 256, 0, stream>>>(h, ei, ei + EDGES, agg, 128);
    k_cast_act<<<(unsigned)(nh / 4 / 256), 256, 0, stream>>>(agg, aggbf, nh);
    k_gemm128<<<(NODES + 63) / 64, 256, 0, stream>>>(
        aggbf, Wt1 + (size_t)l * 128 * 128, bsk + l * 128, h, NODES, 128);
    k_zero<<<1, 256, 0, stream>>>(stats, 256);
    k_bn_stats<<<(NODES + 511) / 512, 256, 0, stream>>>(h, stats, NODES);
    k_bn_apply<<<(unsigned)((nh + 255) / 256), 256, 0, stream>>>(
        h, stats, gammas + l * 128, betas + l * 128, nh, invM);
  }

  // ---- pool + head ----
  k_zero<<<(512 * 128 + 255) / 256, 256, 0, stream>>>(pooled, 512 * 128);
  k_zero<<<2, 256, 0, stream>>>(cnt, 512);
  k_pool<<<(NODES * 32 + 255) / 256, 256, 0, stream>>>(h, batch, pooled, cnt);
  k_comb<<<(512 * 160 + 255) / 256, 256, 0, stream>>>(pooled, cnt, gfeat, comb);
  k_dense<<<(512 * 256 + 255) / 256, 256, 0, stream>>>(comb, Wm1, bm1, z1, 512, 160, 256);
  k_bn_col<<<256, 256, 0, stream>>>(z1, gm1, btm1, 512, 256);
  k_dense<<<(512 * 128 + 255) / 256, 256, 0, stream>>>(z1, Wm2, bm2, z2, 512, 256, 128);
  k_bn_col<<<128, 256, 0, stream>>>(z2, gm2, btm2, 512, 128);
  k_out<<<2, 256, 0, stream>>>(z2, Wm3, bm3, out);
}
